// HistoryRGCN_83932250898807
// MI455X (gfx1250) — matrix operand order repaired and checked by reading, not compiled, not run on hardware
//
#include <hip/hip_runtime.h>
#include <stdint.h>

// CDNA5 WMMA fragment types (wave32)
typedef __attribute__((ext_vector_type(16))) __bf16 vbf16; // A/B: 16 bf16/lane = 8 VGPRs
typedef __attribute__((ext_vector_type(2)))  __bf16 vbf2;
typedef __attribute__((ext_vector_type(8)))  float  v8f;   // C/D: 8 f32/lane

#define NUM_REL 8
#define IN_CH   128
#define HID_CH  128
#define TILE_N  16
#define THREADS 256
#define SPAD    132   // padded fp32 row stride -> conflict-free row-parallel reads

// ---------------------------------------------------------------------------
// Prep kernel: convert fp32 weights [8][128][128] -> bf16 in d_ws, swizzled to
// B-fragment order so a lane's 16-element fragment is 32 contiguous bytes:
//   dst = r*16384 + (((kt*8 + n>>4)*2 + bhi)*16 + (n&15))*16 + tt
//   with kt=k>>5, kk=k&31, bhi=kk>>4, tt=kk&15        (contiguous k order, same as SA)
// ---------------------------------------------------------------------------
__global__ __launch_bounds__(THREADS)
void prep_weights(const float* __restrict__ weight, __bf16* __restrict__ wout) {
    int flat = blockIdx.x * THREADS + threadIdx.x;      // coalesced fp32 read
    if (flat >= NUM_REL * IN_CH * HID_CH) return;
    int r = flat >> 14;
    int k = (flat >> 7) & 127;
    int n = flat & 127;
    int kt = k >> 5, kk = k & 31;
    int bhi = kk >> 4, tt = kk & 15;
    int d = r * 16384 + ((((kt * 8 + (n >> 4)) * 2 + bhi) * 16 + (n & 15)) * 16) + tt;
    wout[d] = (__bf16)weight[flat];
}

// ---------------------------------------------------------------------------
// Main kernel: one workgroup = 16 dst nodes, 256 threads = 8 waves.
//   1) scatter-add aggregation  S[rel][node][128]  (fp32, ds_add_f32)
//   2) one-shot convert S -> SA (bf16, A-fragment-swizzled)
//   3) out[16,128] = sum_r SA_r @ Wbf16[r]  via v_wmma_f32_16x16x32_bf16
//      (A: 2x ds_load_b128, B: 2x global_load_b128 from L2, no barriers)
// ---------------------------------------------------------------------------
__global__ __launch_bounds__(THREADS)
void rgcn_wmma_kernel(const float* __restrict__ x,
                      const __bf16* __restrict__ wbf,      // prepped weights
                      const float* __restrict__ hist_buf,
                      const int*   __restrict__ ptr,
                      const int*   __restrict__ idx,
                      const int*   __restrict__ etype,
                      const unsigned char* __restrict__ used,
                      const int*   __restrict__ hist_map,
                      const int*   __restrict__ hist_size_p,
                      float*       __restrict__ out,
                      int num_node)
{
    __shared__ float  S [NUM_REL * TILE_N * SPAD];   // 66 KB fp32 aggregation
    __shared__ __bf16 SA[NUM_REL * TILE_N * IN_CH];  // 32 KB bf16 A fragments

    const int tid   = threadIdx.x;
    const int wave  = tid >> 5;
    const int lane  = tid & 31;
    const int hi    = lane >> 4;
    const int l16   = lane & 15;
    const int node0 = blockIdx.x * TILE_N;

    // ---- zero aggregation buffer ------------------------------------------
    {
        float4* S4 = (float4*)S;
        const int n4 = (NUM_REL * TILE_N * SPAD) / 4;
        for (int i = tid; i < n4; i += THREADS)
            S4[i] = make_float4(0.f, 0.f, 0.f, 0.f);
    }
    __syncthreads();

    // ---- phase 1: per-(node,rel) aggregation (gather + ds_add_f32) ---------
    for (int s = 0; s < 32; ++s) {
        int e_local = wave * 32 + s;
        int ln   = e_local >> 4;
        int j    = e_local & 15;
        int node = node0 + ln;
        if (node < num_node) {
            int ebase = ptr[node];
            int dg    = ptr[node + 1] - ebase;
            if (j < dg) {
                int e   = ebase + j;
                int sv  = idx[e];
                int rel = used[sv] ? etype[e] : NUM_REL;   // unused src dropped
                if (rel < NUM_REL) {
                    const float4* xr = (const float4*)(x + (size_t)sv * IN_CH);
                    float4 v = xr[lane];
                    float* dst = &S[(rel * TILE_N + ln) * SPAD + lane * 4];
                    atomicAdd(dst + 0, v.x);
                    atomicAdd(dst + 1, v.y);
                    atomicAdd(dst + 2, v.z);
                    atomicAdd(dst + 3, v.w);
                }
            }
        }
    }
    __syncthreads();

    // ---- phase 2: one-shot S -> SA convert, A-fragment swizzle -------------
    // SA flat = ((r*4+kt)*2+hi_a)*256 + row*16 + j ; element j of the A frag
    // maps to K = kt*32 + hi_a*16 + j.
    for (int p = tid; p < (NUM_REL * TILE_N * IN_CH) / 2; p += THREADS) {
        int e0   = 2 * p;
        int j0   = e0 & 15;                       // even
        int row  = (e0 >> 4) & 15;
        int seg  = e0 >> 8;                       // (r*4+kt)*2 + hi_a
        int hia  = seg & 1;
        int ktq  = (seg >> 1) & 3;
        int rr   = seg >> 3;
        int k0   = ktq * 32 + hia * 16 + j0;
        const float2 fv = *(const float2*)&S[(rr * TILE_N + row) * SPAD + k0];
        vbf2 pv; pv[0] = (__bf16)fv.x; pv[1] = (__bf16)fv.y;
        *(vbf2*)&SA[e0] = pv;
    }
    __syncthreads();

    // ---- phase 3: WMMA GEMM, no barriers -----------------------------------
    v8f c = {};
    const int colW = wave;                        // columns [16w, 16w+16)
    for (int r = 0; r < NUM_REL; ++r) {
        #pragma unroll
        for (int kt = 0; kt < IN_CH / 32; ++kt) {
            int abase = ((r * 8 + kt * 2 + hi) * 16 + l16) * 16;
            vbf16 a = *(const vbf16*)&SA[abase];                 // 2x ds_load_b128
            int gbase = r * 16384 + ((kt * 8 + colW) * 2 + hi) * 256 + l16 * 16;
            vbf16 b = *(const vbf16*)&wbf[gbase];                // 2x global_load_b128
            c = __builtin_amdgcn_wmma_f32_16x16x32_bf16(
                    false, a, false, b, (short)0, c, false, false);
        }
    }

    // ---- epilogue: degree norm + history overwrite + store -----------------
    int hsize = hist_size_p[0];
    #pragma unroll
    for (int i = 0; i < 8; ++i) {
        int M    = i + (hi ? 8 : 0);
        int node = node0 + M;
        if (node < num_node) {
            int col = colW * 16 + l16;
            int dg  = ptr[node + 1] - ptr[node];
            float v = c[i] / (float)(dg > 0 ? dg : 1);
            if (hsize > 0 && hist_map[node] != -1)
                v = hist_buf[(size_t)node * HID_CH + col];
            out[(size_t)node * HID_CH + col] = v;
        }
    }
}

extern "C" void kernel_launch(void* const* d_in, const int* in_sizes, int n_in,
                              void* d_out, int out_size, void* d_ws, size_t ws_size,
                              hipStream_t stream) {
    const float*         x   = (const float*)d_in[0];
    const float*         w   = (const float*)d_in[1];
    const float*         hb  = (const float*)d_in[2];
    const int*           ptr = (const int*)d_in[3];
    const int*           idx = (const int*)d_in[4];
    const int*           et  = (const int*)d_in[5];
    const unsigned char* um  = (const unsigned char*)d_in[6];
    const int*           hm  = (const int*)d_in[7];
    const int*           hs  = (const int*)d_in[8];
    float*               out = (float*)d_out;

    int num_node = in_sizes[7];
    __bf16* wbf = (__bf16*)d_ws;                 // 256 KB bf16 swizzled weights

    int wtot = NUM_REL * IN_CH * HID_CH;
    hipLaunchKernelGGL(prep_weights, dim3((wtot + THREADS - 1) / THREADS),
                       dim3(THREADS), 0, stream, w, wbf);

    int grid = (num_node + TILE_N - 1) / TILE_N;
    hipLaunchKernelGGL(rgcn_wmma_kernel, dim3(grid), dim3(THREADS), 0, stream,
                       x, wbf, hb, ptr, idx, et, um, hm, hs, out, num_node);
}
